// PointerGeneratorDecoder_52862457479412
// MI455X (gfx1250) — matrix operand order repaired and checked by reading, not compiled, not run on hardware
//
#include <hip/hip_runtime.h>

// ---------------- problem constants ----------------
constexpr int B_ = 32, T_ = 64, S_ = 400, H_ = 256, E_ = 128, V_ = 50000;
constexpr size_t OUT_VD   = 0;
constexpr size_t OUT_ATTN = (size_t)B_ * T_ * V_;                 // 102,400,000
constexpr size_t OUT_PGEN = OUT_ATTN + (size_t)B_ * T_ * S_;      // +819,200
constexpr size_t OUT_H    = OUT_PGEN + (size_t)B_ * T_;           // +2,048
constexpr size_t OUT_C    = OUT_H + (size_t)B_ * H_;              // +8,192

typedef unsigned short u16;
typedef __attribute__((ext_vector_type(4)))  unsigned int v4u;
typedef __attribute__((ext_vector_type(8)))  unsigned int v8u;
typedef __attribute__((ext_vector_type(16))) __bf16       v16bf;
typedef __attribute__((ext_vector_type(8)))  float        v8f;

// ---------------- bf16 helpers ----------------
__device__ __forceinline__ u16 f2bf(float f) {
  unsigned int u = __builtin_bit_cast(unsigned int, f);
  unsigned int r = u + 0x7FFFu + ((u >> 16) & 1u);   // round-to-nearest-even
  return (u16)(r >> 16);
}
__device__ __forceinline__ float sigmoidf(float x) { return 1.0f / (1.0f + expf(-x)); }

// ---------------- WMMA tile helpers (wave32, 16x16x32 bf16) ----------------
// A-fragment (16x32 bf16, ISA 7.12.2): lane L: M=L%16; lanes<16 hold K {0..7,16..23},
// lanes>=16 hold K {8..15,24..31}, consecutive pairs packed per VGPR.
__device__ __forceinline__ v16bf ldA16(const u16* p) {
  v4u lo = *(const v4u*)p;          // K offsets +0..7   (relative to lane base)
  v4u hi = *(const v4u*)(p + 16);   // K offsets +16..23
  v8u u;
  u[0]=lo[0]; u[1]=lo[1]; u[2]=lo[2]; u[3]=lo[3];
  u[4]=hi[0]; u[5]=hi[1]; u[6]=hi[2]; u[7]=hi[3];
  return __builtin_bit_cast(v16bf, u);
}

struct Acc { v8f c0, c1; };
__device__ __forceinline__ Acc zacc() {
  Acc a;
#pragma unroll
  for (int i = 0; i < 8; ++i) { a.c0[i] = 0.f; a.c1[i] = 0.f; }
  return a;
}

// NT adjacent 32x16 tiles starting at column nbase:  D += A[32,K] * W[N,K]^T.
// A fragments are loaded once per K-step and reused across the NT B tiles
// (raises wmma-per-load density; N is the only reuse axis since M=32).
// A: row-major, row stride lda (bf16).  W: row-major [N,K] (bf16, torch layout).
template <int NT>
__device__ __forceinline__ void gemm_32xN(const u16* A, int lda,
                                          const u16* W, int ldw,
                                          int K, int nbase, Acc* acc) {
  const int lane = threadIdx.x & 31;
  const u16* a0 = A + (size_t)(lane & 15) * lda + ((lane & 16) ? 8 : 0);
  const u16* a1 = a0 + (size_t)16 * lda;
  const u16* wp = W + (size_t)(nbase + (lane & 15)) * ldw + ((lane & 16) ? 8 : 0);
  for (int k = 0; k < K; k += 32) {
    v16bf af0 = ldA16(a0 + k);
    v16bf af1 = ldA16(a1 + k);
#pragma unroll
    for (int i = 0; i < NT; ++i) {
      v16bf bf = ldA16(wp + (size_t)i * 16 * ldw + k);
      acc[i].c0 = __builtin_amdgcn_wmma_f32_16x16x32_bf16(false, af0, false, bf, (short)0, acc[i].c0, false, false);
      acc[i].c1 = __builtin_amdgcn_wmma_f32_16x16x32_bf16(false, af1, false, bf, (short)0, acc[i].c1, false, false);
    }
  }
}

// C layout: VGPR r: lanes 0-15 -> M=r, lanes 16-31 -> M=r+8; N = lane%16.
__device__ __forceinline__ void store_acc(const Acc& a, float* D, int ldd, int nbase,
                                          const float* bias, u16* Dbf, int ldbf) {
  const int lane = threadIdx.x & 31;
  const int n  = nbase + (lane & 15);
  const int mo = (lane & 16) ? 8 : 0;
  const float bv = bias ? bias[n] : 0.0f;
#pragma unroll
  for (int r = 0; r < 8; ++r) {
    float v0 = a.c0[r] + bv;
    float v1 = a.c1[r] + bv;
    if (D)   { D[(mo + r) * ldd + n] = v0;      D[(mo + r + 16) * ldd + n] = v1; }
    if (Dbf) { Dbf[(mo + r) * ldbf + n] = f2bf(v0); Dbf[(mo + r + 16) * ldbf + n] = f2bf(v1); }
  }
}

// ---------------- parameter block ----------------
struct P {
  // inputs
  const int* input_ids; const float* h0; const float* c0;
  const float* enc_out; const unsigned char* mask; const float* emb;
  const float* Wr; const float* br; const float* Wih; const float* Whh;
  const float* bih; const float* bhh; const float* Wh_a; const float* Ws_a;
  const float* bs_a; const float* v_a; const float* wh_p; const float* ws_p;
  const float* wx_p; const float* bx_p; const float* Wvr; const float* bvr;
  const float* Wvo; const float* bvo;
  // output (flat fp32, concatenated)
  float* out;
  // workspace: bf16 weights/activations + fp32 scratch
  u16 *enc_bf, *wha_bf, *wvo_bf, *wre_bf, *wrc_bf, *wih_bf, *whh_bf, *wsa_bf, *wvr_bf, *embs_bf;
  float *enc_feat, *logits, *bsum;
  float *x_f, *gates, *h_f, *c_f, *dec_f, *ctx_f, *ml;
  u16 *x_bf, *h_bf, *state_bf, *ctx_bf, *lstmout_bf, *hv_bf;
};

// ---------------- pre-kernels ----------------
__global__ void k_cvt(const float* s, u16* d, int n) {
  int i = blockIdx.x * 256 + threadIdx.x;
  if (i < n) d[i] = f2bf(s[i]);
}
__global__ void k_cvt_slice(const float* s, u16* d, int rows, int ld, int off, int w) {
  int i = blockIdx.x * 256 + threadIdx.x;
  if (i < rows * w) { int r = i / w, c = i - r * w; d[i] = f2bf(s[(size_t)r * ld + off + c]); }
}
__global__ void k_embs(const int* ids, const float* emb, u16* d) {
  int i = blockIdx.x * 256 + threadIdx.x;
  if (i < B_ * T_ * E_) { int bt = i >> 7, e = i & 127; d[i] = f2bf(emb[(size_t)ids[bt] * E_ + e]); }
}
__global__ void k_init(P p) {
  int i = blockIdx.x * 256 + threadIdx.x;
  if (i < B_ * 2 * H_) p.ctx_bf[i] = 0;                     // bf16(0) == 0x0000
  if (i < B_ * H_) {
    float h = p.h0[i], c = p.c0[i];
    p.h_f[i] = h; p.c_f[i] = c; p.h_bf[i] = f2bf(h);
  }
}
__global__ void k_bsum(P p) {
  int i = blockIdx.x * 256 + threadIdx.x;
  if (i < 4 * H_) p.bsum[i] = p.bih[i] + p.bhh[i];
}
// enc_feat[b,s,:] = encoder_outputs[b,s,:] @ Wh_a^T    (12800 x 512) x (256 x 512)^T
__global__ void k_encfeat(P p) {
  const int w = threadIdx.x >> 5;
  const int mbase = blockIdx.x * 32;
  const u16* A = p.enc_bf + (size_t)mbase * (2 * H_);
  Acc acc[2] = {zacc(), zacc()};
  gemm_32xN<2>(A, 2 * H_, p.wha_bf, 2 * H_, 2 * H_, w * 32, acc);
  store_acc(acc[0], p.enc_feat + (size_t)mbase * H_, H_, w * 32, nullptr, nullptr, 0);
  store_acc(acc[1], p.enc_feat + (size_t)mbase * H_, H_, w * 32 + 16, nullptr, nullptr, 0);
}

// ---------------- per-step kernels ----------------
// K1: the serial cell — x, gates, LSTM pointwise, dec_feat.  One workgroup, 8 waves.
__global__ void k_step_recur(P p, int t) {
  const int w = threadIdx.x >> 5;
  // Phase A: x = emb_t @ Wre^T + ctx @ Wrc^T + br      [32,256]
  {
    Acc acc[2] = {zacc(), zacc()};
    gemm_32xN<2>(p.embs_bf + (size_t)t * E_, T_ * E_, p.wre_bf, E_, E_, w * 32, acc);
    gemm_32xN<2>(p.ctx_bf, 2 * H_, p.wrc_bf, 2 * H_, 2 * H_, w * 32, acc);
    store_acc(acc[0], p.x_f, H_, w * 32, p.br, p.x_bf, H_);
    store_acc(acc[1], p.x_f, H_, w * 32 + 16, p.br, p.x_bf, H_);
  }
  __threadfence_block(); __syncthreads();
  // Phase B: gates = x @ Wih^T + h @ Whh^T + (bih+bhh)  [32,1024]
#pragma unroll
  for (int g = 0; g < 2; ++g) {
    const int nbase = (g * 32 + w * 4) * 16;           // 64 tiles = 2 rounds x 8 waves x 4
    Acc acc[4] = {zacc(), zacc(), zacc(), zacc()};
    gemm_32xN<4>(p.x_bf, H_, p.wih_bf, H_, H_, nbase, acc);
    gemm_32xN<4>(p.h_bf, H_, p.whh_bf, H_, H_, nbase, acc);
#pragma unroll
    for (int i = 0; i < 4; ++i)
      store_acc(acc[i], p.gates, 4 * H_, nbase + i * 16, p.bsum, nullptr, 0);
  }
  __threadfence_block(); __syncthreads();
  // Phase C: LSTM pointwise
  for (int idx = threadIdx.x; idx < B_ * H_; idx += 256) {
    int b = idx >> 8, j = idx & (H_ - 1);
    const float* g = p.gates + (size_t)b * 4 * H_;
    float ig = sigmoidf(g[j]), fg = sigmoidf(g[H_ + j]);
    float gg = tanhf(g[2 * H_ + j]), og = sigmoidf(g[3 * H_ + j]);
    float cn = fg * p.c_f[idx] + ig * gg;
    float hn = og * tanhf(cn);
    p.c_f[idx] = cn; p.h_f[idx] = hn;
    u16 hb = f2bf(hn), cb = f2bf(cn);
    p.h_bf[idx] = hb;
    p.state_bf[(size_t)b * 2 * H_ + j] = hb;
    p.state_bf[(size_t)b * 2 * H_ + H_ + j] = cb;
    p.lstmout_bf[(size_t)b * 3 * H_ + j] = hb;
  }
  __threadfence_block(); __syncthreads();
  // Phase D: dec_feat = [h,c] @ Ws_a^T + bs_a          [32,256]
  {
    Acc acc[2] = {zacc(), zacc()};
    gemm_32xN<2>(p.state_bf, 2 * H_, p.wsa_bf, 2 * H_, 2 * H_, w * 32, acc);
    store_acc(acc[0], p.dec_f, H_, w * 32, p.bs_a, nullptr, 0);
    store_acc(acc[1], p.dec_f, H_, w * 32 + 16, p.bs_a, nullptr, 0);
  }
}

// K2: Bahdanau attention + context + pgen.  One block per batch row (L2-resident streams).
__global__ void k_step_attn(P p, int t) {
  __shared__ float sdec[H_], sva[H_], ssc[S_], red[256];
  const int b = blockIdx.x, tid = threadIdx.x;
  sdec[tid] = p.dec_f[(size_t)b * H_ + tid];
  sva[tid]  = p.v_a[tid];
  __syncthreads();
  for (int s = tid; s < S_; s += 256) {
    const float* ef = p.enc_feat + (size_t)(b * S_ + s) * H_;
    float acc = 0.f;
    for (int h = 0; h < H_; ++h) acc += sva[h] * tanhf(ef[h] + sdec[h]);
    if (p.mask[b * S_ + s]) acc = -__builtin_inff();
    ssc[s] = acc;
  }
  __syncthreads();
  // softmax over S
  float m = -__builtin_inff();
  for (int s = tid; s < S_; s += 256) m = fmaxf(m, ssc[s]);
  red[tid] = m; __syncthreads();
  for (int off = 128; off > 0; off >>= 1) { if (tid < off) red[tid] = fmaxf(red[tid], red[tid + off]); __syncthreads(); }
  m = red[0]; __syncthreads();
  float sum = 0.f;
  for (int s = tid; s < S_; s += 256) { float e = expf(ssc[s] - m); ssc[s] = e; sum += e; }
  red[tid] = sum; __syncthreads();
  for (int off = 128; off > 0; off >>= 1) { if (tid < off) red[tid] += red[tid + off]; __syncthreads(); }
  const float inv = 1.0f / red[0];
  __syncthreads();
  for (int s = tid; s < S_; s += 256) {
    float a = ssc[s] * inv; ssc[s] = a;
    p.out[OUT_ATTN + (size_t)(b * T_ + t) * S_ + s] = a;
  }
  __syncthreads();
  // ctx_new = attn @ encoder_outputs   [2H]
  for (int k = tid; k < 2 * H_; k += 256) {
    const float* eo = p.enc_out + (size_t)b * S_ * (2 * H_) + k;
    float acc = 0.f;
    for (int s = 0; s < S_; ++s) acc += ssc[s] * eo[(size_t)s * (2 * H_)];
    p.ctx_f[(size_t)b * 2 * H_ + k] = acc;
    u16 cb = f2bf(acc);
    p.ctx_bf[(size_t)b * 2 * H_ + k] = cb;
    p.lstmout_bf[(size_t)b * 3 * H_ + H_ + k] = cb;
  }
  // pgen = sigmoid(ctx·wh_p + [h,c]·ws_p + x·wx_p + bx_p)
  float part = 0.f;
  for (int k = tid; k < 2 * H_; k += 256) {
    part += p.ctx_f[(size_t)b * 2 * H_ + k] * p.wh_p[k];
    float sc = (k < H_) ? p.h_f[(size_t)b * H_ + k] : p.c_f[(size_t)b * H_ + (k - H_)];
    part += sc * p.ws_p[k];
    if (k < H_) part += p.x_f[(size_t)b * H_ + k] * p.wx_p[k];
  }
  __syncthreads();
  red[tid] = part; __syncthreads();
  for (int off = 128; off > 0; off >>= 1) { if (tid < off) red[tid] += red[tid + off]; __syncthreads(); }
  if (tid == 0) p.out[OUT_PGEN + (size_t)b * T_ + t] = sigmoidf(red[0] + p.bx_p[0]);
}

// K3pre: hv = [h,ctx] @ Wvr^T + bvr   [32,256] (bf16 for the vocab GEMM)
__global__ void k_step_hv(P p) {
  const int w = threadIdx.x >> 5;
  Acc acc[2] = {zacc(), zacc()};
  gemm_32xN<2>(p.lstmout_bf, 3 * H_, p.wvr_bf, 3 * H_, 3 * H_, w * 32, acc);
  store_acc(acc[0], nullptr, 0, w * 32, p.bvr, p.hv_bf, H_);
  store_acc(acc[1], nullptr, 0, w * 32 + 16, p.bvr, p.hv_bf, H_);
}

// K3a: logits[32,V] = hv @ Wvo^T + bvo.
// 125 blocks x 5 waves; each wave holds 5 N-tiles (80 cols) so A fragments are
// loaded once per K-step and reused -> 1.4 b128 loads per wmma.
__global__ void k_step_vgemm(P p) {
  const int w = threadIdx.x >> 5;
  const int nbase = blockIdx.x * 400 + w * 80;
  __builtin_prefetch(p.wvo_bf + (size_t)(nbase + 400) * H_, 0, 1);  // global_prefetch_b8
  Acc acc[5] = {zacc(), zacc(), zacc(), zacc(), zacc()};
  gemm_32xN<5>(p.hv_bf, H_, p.wvo_bf, H_, H_, nbase, acc);
#pragma unroll
  for (int i = 0; i < 5; ++i)
    store_acc(acc[i], p.logits, V_, nbase + i * 16, p.bvo, nullptr, 0);
}

// K3b: per-row softmax stats (max, 1/sum) over V
__global__ void k_step_stats(P p) {
  __shared__ float red[256];
  const int b = blockIdx.x, tid = threadIdx.x;
  const float* lg = p.logits + (size_t)b * V_;
  float m = -__builtin_inff();
  for (int v = tid; v < V_; v += 256) m = fmaxf(m, lg[v]);
  red[tid] = m; __syncthreads();
  for (int off = 128; off > 0; off >>= 1) { if (tid < off) red[tid] = fmaxf(red[tid], red[tid + off]); __syncthreads(); }
  m = red[0]; __syncthreads();
  float s = 0.f;
  for (int v = tid; v < V_; v += 256) s += expf(lg[v] - m);
  red[tid] = s; __syncthreads();
  for (int off = 128; off > 0; off >>= 1) { if (tid < off) red[tid] += red[tid + off]; __syncthreads(); }
  if (tid == 0) { p.ml[b] = m; p.ml[B_ + b] = 1.0f / red[0]; }
}

// K3c: vd = exp(logit - m) / sum -> vocab_dists[b,t,:]
__global__ void k_step_norm(P p, int t) {
  int idx = blockIdx.x * 256 + threadIdx.x;
  if (idx >= B_ * V_) return;
  int b = idx / V_, v = idx - b * V_;
  float val = expf(p.logits[idx] - p.ml[b]) * p.ml[B_ + b];
  p.out[OUT_VD + (size_t)(b * T_ + t) * V_ + v] = val;
}

__global__ void k_final(P p) {
  int i = blockIdx.x * 256 + threadIdx.x;
  if (i < B_ * H_) { p.out[OUT_H + i] = p.h_f[i]; p.out[OUT_C + i] = p.c_f[i]; }
}

// ---------------- host ----------------
static size_t g_off;
static void* carve(void* base, size_t bytes) {
  void* p = (char*)base + g_off;
  g_off = (g_off + bytes + 255) & ~(size_t)255;
  return p;
}

extern "C" void kernel_launch(void* const* d_in, const int* in_sizes, int n_in,
                              void* d_out, int out_size, void* d_ws, size_t ws_size,
                              hipStream_t stream) {
  (void)in_sizes; (void)n_in; (void)out_size; (void)ws_size;
  P p{};
  p.input_ids = (const int*)d_in[0];
  p.h0   = (const float*)d_in[1];   p.c0   = (const float*)d_in[2];
  p.enc_out = (const float*)d_in[3];
  p.mask = (const unsigned char*)d_in[4];
  p.emb  = (const float*)d_in[5];
  p.Wr   = (const float*)d_in[6];   p.br   = (const float*)d_in[7];
  p.Wih  = (const float*)d_in[8];   p.Whh  = (const float*)d_in[9];
  p.bih  = (const float*)d_in[10];  p.bhh  = (const float*)d_in[11];
  p.Wh_a = (const float*)d_in[12];  p.Ws_a = (const float*)d_in[13];
  p.bs_a = (const float*)d_in[14];  p.v_a  = (const float*)d_in[15];
  p.wh_p = (const float*)d_in[16];  p.ws_p = (const float*)d_in[17];
  p.wx_p = (const float*)d_in[18];  p.bx_p = (const float*)d_in[19];
  p.Wvr  = (const float*)d_in[20];  p.bvr  = (const float*)d_in[21];
  p.Wvo  = (const float*)d_in[22];  p.bvo  = (const float*)d_in[23];
  p.out  = (float*)d_out;

  g_off = 0;
  p.enc_bf  = (u16*)carve(d_ws, (size_t)B_ * S_ * 2 * H_ * 2);
  p.wha_bf  = (u16*)carve(d_ws, (size_t)H_ * 2 * H_ * 2);
  p.wvo_bf  = (u16*)carve(d_ws, (size_t)V_ * H_ * 2);
  p.wre_bf  = (u16*)carve(d_ws, (size_t)H_ * E_ * 2);
  p.wrc_bf  = (u16*)carve(d_ws, (size_t)H_ * 2 * H_ * 2);
  p.wih_bf  = (u16*)carve(d_ws, (size_t)4 * H_ * H_ * 2);
  p.whh_bf  = (u16*)carve(d_ws, (size_t)4 * H_ * H_ * 2);
  p.wsa_bf  = (u16*)carve(d_ws, (size_t)H_ * 2 * H_ * 2);
  p.wvr_bf  = (u16*)carve(d_ws, (size_t)H_ * 3 * H_ * 2);
  p.embs_bf = (u16*)carve(d_ws, (size_t)B_ * T_ * E_ * 2);
  p.enc_feat = (float*)carve(d_ws, (size_t)B_ * S_ * H_ * 4);
  p.logits   = (float*)carve(d_ws, (size_t)B_ * V_ * 4);
  p.bsum     = (float*)carve(d_ws, (size_t)4 * H_ * 4);
  p.x_f      = (float*)carve(d_ws, (size_t)B_ * H_ * 4);
  p.gates    = (float*)carve(d_ws, (size_t)B_ * 4 * H_ * 4);
  p.h_f      = (float*)carve(d_ws, (size_t)B_ * H_ * 4);
  p.c_f      = (float*)carve(d_ws, (size_t)B_ * H_ * 4);
  p.dec_f    = (float*)carve(d_ws, (size_t)B_ * H_ * 4);
  p.ctx_f    = (float*)carve(d_ws, (size_t)B_ * 2 * H_ * 4);
  p.ml       = (float*)carve(d_ws, (size_t)2 * B_ * 4);
  p.x_bf       = (u16*)carve(d_ws, (size_t)B_ * H_ * 2);
  p.h_bf       = (u16*)carve(d_ws, (size_t)B_ * H_ * 2);
  p.state_bf   = (u16*)carve(d_ws, (size_t)B_ * 2 * H_ * 2);
  p.ctx_bf     = (u16*)carve(d_ws, (size_t)B_ * 2 * H_ * 2);
  p.lstmout_bf = (u16*)carve(d_ws, (size_t)B_ * 3 * H_ * 2);
  p.hv_bf      = (u16*)carve(d_ws, (size_t)B_ * H_ * 2);

  auto grid = [](int n) { return (n + 255) / 256; };

  // weight / input preprocessing (bf16)
  k_cvt<<<grid(B_ * S_ * 2 * H_), 256, 0, stream>>>(p.enc_out, p.enc_bf, B_ * S_ * 2 * H_);
  k_cvt<<<grid(H_ * 2 * H_), 256, 0, stream>>>(p.Wh_a, p.wha_bf, H_ * 2 * H_);
  k_cvt<<<grid(V_ * H_), 256, 0, stream>>>(p.Wvo, p.wvo_bf, V_ * H_);
  k_cvt<<<grid(4 * H_ * H_), 256, 0, stream>>>(p.Wih, p.wih_bf, 4 * H_ * H_);
  k_cvt<<<grid(4 * H_ * H_), 256, 0, stream>>>(p.Whh, p.whh_bf, 4 * H_ * H_);
  k_cvt<<<grid(H_ * 2 * H_), 256, 0, stream>>>(p.Ws_a, p.wsa_bf, H_ * 2 * H_);
  k_cvt<<<grid(H_ * 3 * H_), 256, 0, stream>>>(p.Wvr, p.wvr_bf, H_ * 3 * H_);
  k_cvt_slice<<<grid(H_ * E_), 256, 0, stream>>>(p.Wr, p.wre_bf, H_, 2 * H_ + E_, 0, E_);
  k_cvt_slice<<<grid(H_ * 2 * H_), 256, 0, stream>>>(p.Wr, p.wrc_bf, H_, 2 * H_ + E_, E_, 2 * H_);
  k_embs<<<grid(B_ * T_ * E_), 256, 0, stream>>>(p.input_ids, p.emb, p.embs_bf);
  k_init<<<grid(B_ * 2 * H_), 256, 0, stream>>>(p);
  k_bsum<<<grid(4 * H_), 256, 0, stream>>>(p);
  // hoisted encoder features (WMMA GEMM, 400 blocks)
  k_encfeat<<<B_ * S_ / 32, 256, 0, stream>>>(p);

  // decode loop — serial chain on one WGP, wide kernels for attention + vocab
  for (int t = 0; t < T_; ++t) {
    k_step_recur<<<1, 256, 0, stream>>>(p, t);
    k_step_attn<<<B_, 256, 0, stream>>>(p, t);
    k_step_hv<<<1, 256, 0, stream>>>(p);
    k_step_vgemm<<<V_ / 400, 160, 0, stream>>>(p);
    k_step_stats<<<B_, 256, 0, stream>>>(p);
    k_step_norm<<<grid(B_ * V_), 256, 0, stream>>>(p, t);
  }
  k_final<<<grid(B_ * H_), 256, 0, stream>>>(p);
}
